// MultiheadSelfAttention_75213467287622
// MI455X (gfx1250) — matrix operand order repaired and checked by reading, not compiled, not run on hardware
//
#include <hip/hip_runtime.h>
#include <hip/hip_bf16.h>
#include <stdint.h>

// ---------------------------------------------------------------------------
// Multihead self-attention with RoPE for MI455X (gfx1250, wave32, WMMA).
// All matmuls use v_wmma_f32_16x16x32_f16 (f16 inputs, f32 accumulate).
// Attention K-tiles are staged into LDS by the Tensor Data Mover with
// double buffering (tensor_load_to_lds + s_wait_tensorcnt).
// ---------------------------------------------------------------------------

#define D_MODEL   1024
#define NUM_HEADS 16
#define D_K       64
#define BATCH     2
#define SEQ       2048
#define M_TOT     (BATCH * SEQ)          // 4096 rows of X
#define LOG_THETA 9.210340371976184f     // ln(10000)

typedef __attribute__((ext_vector_type(16))) _Float16 v16h;
typedef __attribute__((ext_vector_type(8)))  float    v8f;
typedef __attribute__((ext_vector_type(4)))  unsigned int tdm_g0_t;
typedef __attribute__((ext_vector_type(8)))  int          tdm_g1_t;
typedef __attribute__((ext_vector_type(4)))  int          tdm_g2_t;
typedef __attribute__((ext_vector_type(8)))  int          tdm_g4_t;

__device__ __forceinline__ v8f wmma_f16(v16h a, v16h b, v8f c) {
  // (neg_a, A, neg_b, B, c_mod, C, reuse_a, reuse_b)
  return __builtin_amdgcn_wmma_f32_16x16x32_f16(false, a, false, b, (short)0, c,
                                                false, false);
}

// TDM: DMA one 32-key x 64-d f16 tile (row-major, row stride 64) into LDS.
// D# per CDNA5 ISA (2D tensor, groups 2/3 zero).
// 6-arg builtin form (clang-23 / therock-10.0 toolchain).
__device__ __forceinline__ void tdm_load_k_tile(unsigned lds_off,
                                                const _Float16* gptr) {
  unsigned long long ga = (unsigned long long)(uintptr_t)gptr;
  tdm_g0_t g0;
  g0[0] = 1u;                                        // count=1 (valid user D#)
  g0[1] = lds_off;                                   // lds_addr (bytes)
  g0[2] = (unsigned)(ga & 0xFFFFFFFFu);              // global_addr[31:0]
  g0[3] = (unsigned)((ga >> 32) & 0x01FFFFFFu) | (2u << 30);  // addr hi | type=2
  tdm_g1_t g1;
  g1[0] = (int)(1u << 16);        // workgroup_mask=0, data_size=1 (2 bytes)
  g1[1] = (int)(64u << 16);       // tensor_dim0 = 64 (bits 79:48 lo)
  g1[2] = (int)(32u << 16);       // tensor_dim0 hi=0, tensor_dim1 = 32
  g1[3] = (int)(64u << 16);       // tensor_dim1 hi=0, tile_dim0 = 64
  g1[4] = 32;                     // tile_dim1 = 32, tile_dim2 = 0
  g1[5] = 64;                     // tensor_dim0_stride = 64 (lo 32)
  g1[6] = 0;                      // stride hi, tensor_dim1_stride lo
  g1[7] = 0;
  tdm_g2_t z4 = {0, 0, 0, 0};
  tdm_g4_t z8 = {0, 0, 0, 0, 0, 0, 0, 0};
  __builtin_amdgcn_tensor_load_to_lds(g0, g1, z4, z4, z8, 0);
}

// ---------------------------------------------------------------------------
// fp32 -> fp16 conversion
// ---------------------------------------------------------------------------
__global__ __launch_bounds__(256) void f32_to_f16_kernel(
    const float* __restrict__ src, _Float16* __restrict__ dst, long n) {
  long t = (long)blockIdx.x * blockDim.x + threadIdx.x;
  if (t < n) dst[t] = (_Float16)src[t];
}

// ---------------------------------------------------------------------------
// C[M x N] (f32) = A[M x K] (f16) * W[N x K]^T (f16)
// One wave computes a 32 x 64 strip: 2 A fragments x 4 B fragments -> 8 WMMA
// per K-step, 12 b128 loads -> 1.5 loads/WMMA. Block = 8 waves = 256 x 64.
// ---------------------------------------------------------------------------
__global__ __launch_bounds__(256) void gemm_xwT_kernel(
    const _Float16* __restrict__ A, const _Float16* __restrict__ W,
    float* __restrict__ C, int M, int N, int K) {
  const int lane = threadIdx.x & 31;
  const int wave = threadIdx.x >> 5;
  const int hf   = lane >> 4;        // K-group select (wave half)
  const int nl   = lane & 15;

  const int m0 = blockIdx.y * 256 + wave * 32;
  const int n0 = blockIdx.x * 64;

  v8f acc[2][4] = {{{}, {}, {}, {}}, {{}, {}, {}, {}}};

  const _Float16* __restrict__ arow0 = A + (long)(m0 + nl) * K;
  const _Float16* __restrict__ arow1 = A + (long)(m0 + 16 + nl) * K;

  for (int k0 = 0; k0 < K; k0 += 32) {
    if (k0 + 32 < K) {
      __builtin_prefetch(arow0 + k0 + 32, 0, 1);   // global_prefetch_b8
      __builtin_prefetch(arow1 + k0 + 32, 0, 1);
    }

    v16h a0, a1;
#pragma unroll
    for (int i = 0; i < 8; ++i) {
      const int kk = k0 + hf * 16 + i * 2;
      a0[2 * i]     = arow0[kk];
      a0[2 * i + 1] = arow0[kk + 1];
      a1[2 * i]     = arow1[kk];
      a1[2 * i + 1] = arow1[kk + 1];
    }

#pragma unroll
    for (int jn = 0; jn < 4; ++jn) {
      const _Float16* __restrict__ wrow =
          W + (long)(n0 + jn * 16 + nl) * K + k0 + hf * 16;
      v16h bf;
#pragma unroll
      for (int i = 0; i < 16; ++i) bf[i] = wrow[i];   // 16 contiguous halves
      acc[0][jn] = wmma_f16(a0, bf, acc[0][jn]);
      acc[1][jn] = wmma_f16(a1, bf, acc[1][jn]);
    }
  }

  // C/D layout: lane holds column n0 + lane%16; rows m0 + (lane/16)*8 + r.
#pragma unroll
  for (int t = 0; t < 2; ++t) {
#pragma unroll
    for (int jn = 0; jn < 4; ++jn) {
      float* cp = C + (long)(m0 + t * 16 + hf * 8) * N + n0 + jn * 16 + nl;
#pragma unroll
      for (int r = 0; r < 8; ++r) cp[(long)r * N] = acc[t][jn][r];
    }
  }
}

// ---------------------------------------------------------------------------
// RoPE on Q,K + head-split + f16 conversion.
// Q,K -> [B, H, S, D_K] (row-major, for A-fragments / K-tiles).
// V   -> [B, H, D_K, S] (TRANSPOSED, so attention B-fragments of V are
//        16 contiguous halves -> 2 x global_load_b128).
// ---------------------------------------------------------------------------
__global__ __launch_bounds__(256) void rope_split_kernel(
    const float* __restrict__ Qf, const float* __restrict__ Kf,
    const float* __restrict__ Vf, const int* __restrict__ pos,
    _Float16* __restrict__ Qh, _Float16* __restrict__ Kh,
    _Float16* __restrict__ Vt) {
  const long t = (long)blockIdx.x * blockDim.x + threadIdx.x;
  const int PAIRS = D_MODEL / 2;            // 512 pairs per token
  const long bs = t / PAIRS;
  if (bs >= (long)M_TOT) return;
  const int rem = (int)(t % PAIRS);         // rem = h*32 + i
  const int hh = rem >> 5;
  const int i  = rem & 31;                  // pair index within head
  const int s  = (int)(bs % SEQ);
  const int b  = (int)(bs / SEQ);

  const float p   = (float)pos[s];
  const float inv = __expf(-(float)(2 * i) * (LOG_THETA / (float)D_K));
  float sn, cs;
  __sincosf(p * inv, &sn, &cs);

  const long src = bs * D_MODEL + (long)hh * D_K + 2 * i;
  const float q1 = Qf[src], q2 = Qf[src + 1];
  const float k1 = Kf[src], k2 = Kf[src + 1];

  const long dst = (((long)b * NUM_HEADS + hh) * SEQ + s) * D_K + 2 * i;
  Qh[dst]     = (_Float16)(q1 * cs - q2 * sn);
  Qh[dst + 1] = (_Float16)(q1 * sn + q2 * cs);
  Kh[dst]     = (_Float16)(k1 * cs - k2 * sn);
  Kh[dst + 1] = (_Float16)(k1 * sn + k2 * cs);

  const long vdst = (((long)b * NUM_HEADS + hh) * D_K + 2 * i) * SEQ + s;
  Vt[vdst]       = (_Float16)Vf[src];
  Vt[vdst + SEQ] = (_Float16)Vf[src + 1];
}

// ---------------------------------------------------------------------------
// Causal flash attention, one wave per (b, h, 16-query tile).
// K-tiles (32 x 64 f16) DMA'd into LDS by the TDM, double buffered: the next
// tile's tensor_load_to_lds is issued right after s_wait_tensorcnt(0), so DMA
// of tile t+1 overlaps the WMMAs of tile t. V read directly (transposed).
// ---------------------------------------------------------------------------
#define PSTR 40   // padded LDS row stride in halves (16x32 P tile)

__global__ __launch_bounds__(32) void attn_kernel(
    const _Float16* __restrict__ Qh, const _Float16* __restrict__ Kh,
    const _Float16* __restrict__ Vt, float* __restrict__ Out) {
  __shared__ _Float16 Kl[2][32 * D_K];     // double-buffered K tile (8 KB)
  __shared__ _Float16 Pl[16 * PSTR];       // probability staging tile

  const int lane = threadIdx.x & 31;
  const int hf   = lane >> 4;
  const int nl   = lane & 15;

  const int q0 = blockIdx.x * 16;
  const int h  = blockIdx.y;
  const int b  = blockIdx.z;

  const long base = ((long)(b * NUM_HEADS + h) * SEQ) * D_K;
  const _Float16* __restrict__ Qb = Qh + base + (long)q0 * D_K;
  const _Float16* __restrict__ Kb = Kh + base;
  const _Float16* __restrict__ Vb = Vt + base;   // [D_K, SEQ] for this (b,h)

  // Load the two Q A-fragments (d 0..31 and 32..63) once.
  v16h aq0, aq1;
  {
    const _Float16* qrow = Qb + (long)nl * D_K;
#pragma unroll
    for (int i = 0; i < 8; ++i) {
      const int kk = hf * 16 + i * 2;
      aq0[2 * i]     = qrow[kk];
      aq0[2 * i + 1] = qrow[kk + 1];
      aq1[2 * i]     = qrow[32 + kk];
      aq1[2 * i + 1] = qrow[32 + kk + 1];
    }
  }

  float mrow[8], lrow[8];
  v8f acc[4] = {{}, {}, {}, {}};
#pragma unroll
  for (int r = 0; r < 8; ++r) { mrow[r] = -__builtin_inff(); lrow[r] = 0.0f; }

  const float scale = 0.125f;  // 1/sqrt(64)
  const int qmax = q0 + 15;
  const int nt = qmax / 32 + 1;  // number of 32-key tiles

  const unsigned klds0 = (unsigned)(uintptr_t)&Kl[0][0];
  const unsigned klds1 = (unsigned)(uintptr_t)&Kl[1][0];

  // Preload tile 0.
  tdm_load_k_tile(klds0, Kb);

  for (int t = 0; t < nt; ++t) {
    const int kt0 = t * 32;
    __builtin_amdgcn_s_wait_tensorcnt(0);    // current tile resident in LDS
    __asm__ volatile("" ::: "memory");
    if (t + 1 < nt)                          // overlap next DMA with compute
      tdm_load_k_tile((t & 1) ? klds0 : klds1, Kb + (long)(kt0 + 32) * D_K);

    const _Float16* __restrict__ Kt = &Kl[t & 1][0];

    // ---- scores for keys [kt0, kt0+16) and [kt0+16, kt0+32) ----
    v8f sc0 = {}, sc1 = {};
    {
      const _Float16* kr0 = Kt + (nl) * D_K + hf * 16;
      const _Float16* kr1 = Kt + (16 + nl) * D_K + hf * 16;
      v16h bk;
#pragma unroll
      for (int i = 0; i < 16; ++i) bk[i] = kr0[i];
      sc0 = wmma_f16(aq0, bk, sc0);
#pragma unroll
      for (int i = 0; i < 16; ++i) bk[i] = kr0[32 + i];
      sc0 = wmma_f16(aq1, bk, sc0);
#pragma unroll
      for (int i = 0; i < 16; ++i) bk[i] = kr1[i];
      sc1 = wmma_f16(aq0, bk, sc1);
#pragma unroll
      for (int i = 0; i < 16; ++i) bk[i] = kr1[32 + i];
      sc1 = wmma_f16(aq1, bk, sc1);
    }

    // ---- causal mask + online softmax (rows = hf*8 + r, cols = nl) ----
    float p0[8], p1[8];
#pragma unroll
    for (int r = 0; r < 8; ++r) {
      const int q = q0 + hf * 8 + r;
      float s0 = sc0[r] * scale;
      float s1 = sc1[r] * scale;
      if (kt0 + nl > q)      s0 = -__builtin_inff();
      if (kt0 + 16 + nl > q) s1 = -__builtin_inff();

      float mx = fmaxf(s0, s1);
#pragma unroll
      for (int off = 8; off >= 1; off >>= 1)
        mx = fmaxf(mx, __shfl_xor(mx, off, 32));   // reduce within 16-lane half
      mx = fmaxf(mx, mrow[r]);

      const float e0 = __expf(s0 - mx);
      const float e1 = __expf(s1 - mx);
      float sum = e0 + e1;
#pragma unroll
      for (int off = 8; off >= 1; off >>= 1)
        sum += __shfl_xor(sum, off, 32);

      const float rs = __expf(mrow[r] - mx);
      lrow[r] = lrow[r] * rs + sum;
      mrow[r] = mx;
      p0[r] = e0;
      p1[r] = e1;
#pragma unroll
      for (int jn = 0; jn < 4; ++jn) acc[jn][r] *= rs;
    }

    // ---- C-layout probs -> LDS -> A-layout fragment ----
#pragma unroll
    for (int r = 0; r < 8; ++r) {
      Pl[(hf * 8 + r) * PSTR + nl]      = (_Float16)p0[r];
      Pl[(hf * 8 + r) * PSTR + 16 + nl] = (_Float16)p1[r];
    }
    __syncthreads();
    v16h ap;
#pragma unroll
    for (int i = 0; i < 8; ++i) {
      const int kk = hf * 16 + i * 2;
      ap[2 * i]     = Pl[nl * PSTR + kk];
      ap[2 * i + 1] = Pl[nl * PSTR + kk + 1];
    }
    __syncthreads();

    // ---- P (16x32) x V (32x64): 4 WMMAs, contiguous V loads ----
#pragma unroll
    for (int jn = 0; jn < 4; ++jn) {
      const _Float16* vr = Vb + (long)(jn * 16 + nl) * SEQ + kt0 + hf * 16;
      v16h bv;
#pragma unroll
      for (int i = 0; i < 16; ++i) bv[i] = vr[i];   // 16 contiguous halves
      acc[jn] = wmma_f16(ap, bv, acc[jn]);
    }
  }

  // ---- normalize and store to [B, S, D_MODEL] with head interleave ----
#pragma unroll
  for (int jn = 0; jn < 4; ++jn) {
#pragma unroll
    for (int r = 0; r < 8; ++r) {
      const int q = q0 + hf * 8 + r;
      const long o = ((long)b * SEQ + q) * D_MODEL + h * D_K + jn * 16 + nl;
      Out[o] = acc[jn][r] / lrow[r];
    }
  }
}

// ---------------------------------------------------------------------------
// Host-side orchestration
// ---------------------------------------------------------------------------
extern "C" void kernel_launch(void* const* d_in, const int* in_sizes, int n_in,
                              void* d_out, int out_size, void* d_ws, size_t ws_size,
                              hipStream_t stream) {
  (void)in_sizes; (void)n_in; (void)out_size; (void)ws_size;

  const float* x   = (const float*)d_in[0];
  const int*   tp  = (const int*)d_in[1];
  const float* Wq  = (const float*)d_in[2];
  const float* Wk  = (const float*)d_in[3];
  const float* Wv  = (const float*)d_in[4];
  const float* Wo  = (const float*)d_in[5];
  float* out = (float*)d_out;

  const long XN = (long)M_TOT * D_MODEL;       // 4,194,304
  const long WN = (long)D_MODEL * D_MODEL;     // 1,048,576

  // Workspace carve-up: 88 MB total with aliasing.
  char* w = (char*)d_ws;
  _Float16* xh  = (_Float16*)w;                 w += XN * 2;        // 8 MB
  _Float16* wqh = (_Float16*)w;                 w += WN * 2;        // 2 MB
  _Float16* wkh = (_Float16*)w;                 w += WN * 2;
  _Float16* wvh = (_Float16*)w;                 w += WN * 2;
  _Float16* woh = (_Float16*)w;                 w += WN * 2;
  float* qf = (float*)w;                        w += XN * 4;        // 16 MB
  float* kf = (float*)w;                        w += XN * 4;
  float* vf = (float*)w;                        w += XN * 4;
  _Float16* qh = (_Float16*)w;                  w += XN * 2;        // 8 MB
  _Float16* kh = (_Float16*)w;                  w += XN * 2;
  _Float16* vt = (_Float16*)w;                  w += XN * 2;
  float*    ao  = qf;                 // alias: q/k/v f32 dead after RoPE
  _Float16* aoh = (_Float16*)kf;      // alias

  // 1) fp32 -> fp16 conversions
  f32_to_f16_kernel<<<(int)((XN + 255) / 256), 256, 0, stream>>>(x, xh, XN);
  f32_to_f16_kernel<<<(int)((WN + 255) / 256), 256, 0, stream>>>(Wq, wqh, WN);
  f32_to_f16_kernel<<<(int)((WN + 255) / 256), 256, 0, stream>>>(Wk, wkh, WN);
  f32_to_f16_kernel<<<(int)((WN + 255) / 256), 256, 0, stream>>>(Wv, wvh, WN);
  f32_to_f16_kernel<<<(int)((WN + 255) / 256), 256, 0, stream>>>(Wo, woh, WN);

  // 2) Q/K/V projections: C = X * W^T
  dim3 ggrid(D_MODEL / 64, M_TOT / 256);
  gemm_xwT_kernel<<<ggrid, 256, 0, stream>>>(xh, wqh, qf, M_TOT, D_MODEL, D_MODEL);
  gemm_xwT_kernel<<<ggrid, 256, 0, stream>>>(xh, wkh, kf, M_TOT, D_MODEL, D_MODEL);
  gemm_xwT_kernel<<<ggrid, 256, 0, stream>>>(xh, wvh, vf, M_TOT, D_MODEL, D_MODEL);

  // 3) RoPE + head split + f16 (V transposed)
  const long RT = (long)M_TOT * (D_MODEL / 2);
  rope_split_kernel<<<(int)((RT + 255) / 256), 256, 0, stream>>>(
      qf, kf, vf, tp, qh, kh, vt);

  // 4) causal flash attention (TDM-staged K tiles)
  dim3 agrid(SEQ / 16, NUM_HEADS, BATCH);
  attn_kernel<<<agrid, 32, 0, stream>>>(qh, kh, vt, ao);

  // 5) output projection
  f32_to_f16_kernel<<<(int)((XN + 255) / 256), 256, 0, stream>>>(ao, aoh, XN);
  gemm_xwT_kernel<<<ggrid, 256, 0, stream>>>(aoh, woh, out, M_TOT, D_MODEL, D_MODEL);
}
